// Decoder_5076651344146
// MI455X (gfx1250) — matrix operand order repaired and checked by reading, not compiled, not run on hardware
//
#include <hip/hip_runtime.h>

#define WAVES  4
#define B_TOT  32768
#define HDIM   64
#define DDIM   128
#define NSTEP  16

typedef _Float16 v16h __attribute__((ext_vector_type(16)));
typedef _Float16 v8h  __attribute__((ext_vector_type(8)));
typedef float    v8f  __attribute__((ext_vector_type(8)));
typedef float    v4f  __attribute__((ext_vector_type(4)));

__device__ __forceinline__ v16h cat16(v8h lo, v8h hi) {
  return __builtin_shufflevector(lo, hi, 0,1,2,3,4,5,6,7,8,9,10,11,12,13,14,15);
}

__device__ __forceinline__ v8f wmma_f16(v16h a, v16h b, v8f c) {
  // D = A(16x32 f16) * B(32x16 f16) + C(16x16 f32)
  return __builtin_amdgcn_wmma_f32_16x16x32_f16(false, a, false, b, (short)0, c, false, false);
}

// LDS f16 matrix, row stride 64 halves; 16B chunks XOR-swizzled by ((row>>1)&7)
// A-fragment: lane holds row tile*16+(lane&15); halves {kc+sel*8..+7, kc+16+sel*8..+7}
__device__ __forceinline__ v16h load_afrag(const _Float16* base, int tile, int kc, int lane) {
  int row = tile * 16 + (lane & 15);
  int sw  = (row >> 1) & 7;
  int c0  = (kc >> 3) + (lane >> 4);
  int c1  = c0 + 2;
  const _Float16* r = base + row * 64;
  v8h lo = *(const v8h*)(r + ((c0 ^ sw) << 3));
  v8h hi = *(const v8h*)(r + ((c1 ^ sw) << 3));
  return cat16(lo, hi);
}

// B-fragment: lane holds column (batch row) lane&15; halves {kc+sel*8..+7, kc+16+sel*8..+7}
__device__ __forceinline__ v16h load_bfrag(const _Float16* base, int kc, int lane) {
  int row = lane & 15;
  int sw  = (row >> 1) & 7;
  int c0  = (kc >> 3) + (lane >> 4);
  const _Float16* r = base + row * 64;
  v8h lo = *(const v8h*)(r + ((c0 ^ sw) << 3));
  v8h hi = *(const v8h*)(r + (((c0 + 2) ^ sw) << 3));
  return cat16(lo, hi);
}

// store one D tile (8 f32 -> 8 consecutive f16 of row lane&15) into staging buffer
__device__ __forceinline__ void store_row8(_Float16* base, int lane, int j, v8f v) {
  int row = lane & 15;
  int sw  = (row >> 1) & 7;
  int c   = (j << 1) + (lane >> 4);
  v8h h;
  #pragma unroll
  for (int i = 0; i < 8; ++i) h[i] = (_Float16)v[i];
  *(v8h*)(base + row * 64 + ((c ^ sw) << 3)) = h;
}

__device__ __forceinline__ v8f load_bias8(const float* b, int off) {
  v4f lo = *(const v4f*)(b + off);
  v4f hi = *(const v4f*)(b + off + 4);
  return __builtin_shufflevector(lo, hi, 0,1,2,3,4,5,6,7);
}

__device__ __forceinline__ float fast_tanh(float x) {
#if __has_builtin(__builtin_amdgcn_tanhf)
  return __builtin_amdgcn_tanhf(x);          // v_tanh_f32 (CDNA5 TRANS op)
#elif __has_builtin(__builtin_amdgcn_tanh_f32)
  return __builtin_amdgcn_tanh_f32(x);
#else
  return 2.f / (1.f + __expf(-2.f * x)) - 1.f;
#endif
}
__device__ __forceinline__ float fast_sig(float x) { return 0.5f * fast_tanh(0.5f * x) + 0.5f; }

__global__ __launch_bounds__(32 * WAVES) void gru_decoder_wmma(
    const float* __restrict__ z,    const int* __restrict__ nseq,
    const float* __restrict__ W_ih, const float* __restrict__ W_hh,
    const float* __restrict__ b_ih, const float* __restrict__ b_hh,
    const float* __restrict__ W1,   const float* __restrict__ b1,
    const float* __restrict__ W2,   const float* __restrict__ b2,
    float* __restrict__ xout, float* __restrict__ maskout)
{
  // weights in f16, row-major rows = output channels, K = 64 (swizzled chunks)
  __shared__ _Float16 sW[384 * 64];    // [W_ih ; W_hh] rows 0..383
  __shared__ _Float16 sW1t[64 * 64];   // sW1t[n][k] = W1[k][n]
  __shared__ _Float16 sW2t[128 * 64];  // sW2t[n][k] = W2[k][n]
  __shared__ float bRs[64], bZs[64], bIcs[64], bHcs[64], b1s[64], b2s[128];
  __shared__ _Float16 sH[WAVES][32 * 64];  // h   (f16, row-major [32][64], 2 sub-tiles)
  __shared__ _Float16 sM[WAVES][32 * 64];  // MLP hidden

  const int tid  = threadIdx.x;
  const int lane = tid & 31;
  const int w    = tid >> 5;
  const int rowBase = (blockIdx.x * WAVES + w) * 32;   // 32 batch rows per wave
  const int sel8 = (lane >> 4) << 3;

  // ---- cooperative weight staging (f32 -> f16, chunk-swizzled) ----
  for (int idx = tid; idx < 384 * 8; idx += blockDim.x) {
    int row = idx >> 3, c = idx & 7;
    const float* src = (row < 192) ? (W_ih + row * 64 + c * 8)
                                   : (W_hh + (row - 192) * 64 + c * 8);
    v4f a = *(const v4f*)src, b = *(const v4f*)(src + 4);
    v8h hv;
    #pragma unroll
    for (int i = 0; i < 4; ++i) { hv[i] = (_Float16)a[i]; hv[4 + i] = (_Float16)b[i]; }
    int sw = (row >> 1) & 7;
    *(v8h*)(sW + row * 64 + ((c ^ sw) << 3)) = hv;
  }
  for (int idx = tid; idx < 64 * 64; idx += blockDim.x) {
    int nn = idx >> 6, k = idx & 63;
    int sw = (nn >> 1) & 7;
    sW1t[nn * 64 + (((k >> 3) ^ sw) << 3) + (k & 7)] = (_Float16)W1[k * 64 + nn];
  }
  for (int idx = tid; idx < 128 * 64; idx += blockDim.x) {
    int nn = idx >> 6, k = idx & 63;
    int sw = (nn >> 1) & 7;
    sW2t[nn * 64 + (((k >> 3) ^ sw) << 3) + (k & 7)] = (_Float16)W2[k * 128 + nn];
  }
  for (int idx = tid; idx < 64; idx += blockDim.x) {
    bRs[idx]  = b_ih[idx]       + b_hh[idx];
    bZs[idx]  = b_ih[64 + idx]  + b_hh[64 + idx];
    bIcs[idx] = b_ih[128 + idx];
    bHcs[idx] = b_hh[128 + idx];
    b1s[idx]  = b1[idx];
  }
  for (int idx = tid; idx < 128; idx += blockDim.x) b2s[idx] = b2[idx];

  // ---- per-wave state init: sH = f16(z rows) ----
  _Float16* sHw = sH[w]; _Float16* sMw = sM[w];
  for (int idx = lane; idx < 256; idx += 32) {
    int row = idx >> 3, c = idx & 7;      // row 0..31 within wave tile
    const float* zp = z + (size_t)(rowBase + row) * HDIM + c * 8;
    v4f a = *(const v4f*)zp, b = *(const v4f*)(zp + 4);
    v8h hv;
    #pragma unroll
    for (int i = 0; i < 4; ++i) { hv[i] = (_Float16)a[i]; hv[4 + i] = (_Float16)b[i]; }
    int sw = (row >> 1) & 7;
    *(v8h*)(sHw + row * 64 + ((c ^ sw) << 3)) = hv;
  }

  // h in transposed-D register layout per sub-tile: lane holds batch row lane&15,
  // hreg[s][j][v] = h[s*16+row][j*16 + sel8 + v]   (f32 for exact gate math)
  v8f hreg[2][4];
  #pragma unroll
  for (int s = 0; s < 2; ++s) {
    const float* zr = z + (size_t)(rowBase + s * 16 + (lane & 15)) * HDIM + sel8;
    #pragma unroll
    for (int j = 0; j < 4; ++j) {
      v4f a = *(const v4f*)(zr + j * 16);
      v4f b = *(const v4f*)(zr + j * 16 + 4);
      hreg[s][j] = __builtin_shufflevector(a, b, 0,1,2,3,4,5,6,7);
    }
  }
  const int nv0    = nseq[rowBase + (lane & 15)];
  const int nv1    = nseq[rowBase + 16 + (lane & 15)];
  const int nstore = nseq[rowBase + lane];
  __syncthreads();

  // h^T B-fragments (carried across iterations; also serve as MLP1 input)
  v16h bh[2][2];
  #pragma unroll
  for (int s = 0; s < 2; ++s) {
    bh[s][0] = load_bfrag(sHw + s * 1024, 0,  lane);
    bh[s][1] = load_bfrag(sHw + s * 1024, 32, lane);
  }

  #pragma unroll 1
  for (int t = 0; t < NSTEP; ++t) {
    // x == h for t>0 ; x == 0 at t==0 (GEMM side contributes nothing)
    v16h zf;
    #pragma unroll
    for (int i = 0; i < 16; ++i) zf[i] = (_Float16)0.f;
    v16h bxs[2][2];
    #pragma unroll
    for (int s = 0; s < 2; ++s) {
      bxs[s][0] = (t == 0) ? zf : bh[s][0];
      bxs[s][1] = (t == 0) ? zf : bh[s][1];
    }

    #pragma unroll
    for (int j = 0; j < 4; ++j) {
      // r gate: ir+hr fused into one 4-WMMA chain, C seeded with bias
      v8f r2[2], u2[2], c2[2], hc2[2];
      {
        v16h a0 = load_afrag(sW, j,      0, lane), a1 = load_afrag(sW, j,      32, lane);
        v16h a2 = load_afrag(sW, 12 + j, 0, lane), a3 = load_afrag(sW, 12 + j, 32, lane);
        v8f bb = load_bias8(bRs, j * 16 + sel8);
        #pragma unroll
        for (int s = 0; s < 2; ++s) {
          v8f acc = bb;
          acc = wmma_f16(a0, bxs[s][0], acc);
          acc = wmma_f16(a1, bxs[s][1], acc);
          acc = wmma_f16(a2, bh[s][0],  acc);
          acc = wmma_f16(a3, bh[s][1],  acc);
          r2[s] = acc;
        }
      }
      { // z gate: iz+hz fused
        v16h a0 = load_afrag(sW, 4 + j,  0, lane), a1 = load_afrag(sW, 4 + j,  32, lane);
        v16h a2 = load_afrag(sW, 16 + j, 0, lane), a3 = load_afrag(sW, 16 + j, 32, lane);
        v8f bb = load_bias8(bZs, j * 16 + sel8);
        #pragma unroll
        for (int s = 0; s < 2; ++s) {
          v8f acc = bb;
          acc = wmma_f16(a0, bxs[s][0], acc);
          acc = wmma_f16(a1, bxs[s][1], acc);
          acc = wmma_f16(a2, bh[s][0],  acc);
          acc = wmma_f16(a3, bh[s][1],  acc);
          u2[s] = acc;
        }
      }
      { // candidate: ic (x side) and hc (h side) kept separate, biases seeded
        v16h a0 = load_afrag(sW, 8 + j,  0, lane), a1 = load_afrag(sW, 8 + j,  32, lane);
        v16h a2 = load_afrag(sW, 20 + j, 0, lane), a3 = load_afrag(sW, 20 + j, 32, lane);
        v8f bic = load_bias8(bIcs, j * 16 + sel8);
        v8f bhc = load_bias8(bHcs, j * 16 + sel8);
        #pragma unroll
        for (int s = 0; s < 2; ++s) {
          v8f ai = bic;
          ai = wmma_f16(a0, bxs[s][0], ai);
          ai = wmma_f16(a1, bxs[s][1], ai);
          c2[s] = ai;
          v8f ah = bhc;
          ah = wmma_f16(a2, bh[s][0], ah);
          ah = wmma_f16(a3, bh[s][1], ah);
          hc2[s] = ah;
        }
      }
      #pragma unroll
      for (int s = 0; s < 2; ++s) {
        v8f hn;
        #pragma unroll
        for (int i = 0; i < 8; ++i) {
          float r  = fast_sig(r2[s][i]);
          float u  = fast_sig(u2[s][i]);
          float cc = fast_tanh(c2[s][i] + r * hc2[s][i]);
          hn[i] = cc + u * (hreg[s][j][i] - cc);
        }
        hreg[s][j] = hn;
      }
    }
    #pragma unroll
    for (int s = 0; s < 2; ++s)
      #pragma unroll
      for (int j = 0; j < 4; ++j)
        store_row8(sHw + s * 1024, lane, j, hreg[s][j]);   // h_new (== next x)
    __builtin_amdgcn_wave_barrier();       // order LDS stores vs. following loads

    // reload h^T fragments: feed MLP1 now, and next step's gates
    #pragma unroll
    for (int s = 0; s < 2; ++s) {
      bh[s][0] = load_bfrag(sHw + s * 1024, 0,  lane);
      bh[s][1] = load_bfrag(sHw + s * 1024, 32, lane);
    }

    // MLP layer 1: relu(y @ W1 + b1), bias seeded in C
    #pragma unroll
    for (int j = 0; j < 4; ++j) {
      v16h a0 = load_afrag(sW1t, j, 0, lane), a1 = load_afrag(sW1t, j, 32, lane);
      v8f bb = load_bias8(b1s, j * 16 + sel8);
      #pragma unroll
      for (int s = 0; s < 2; ++s) {
        v8f m = bb;
        m = wmma_f16(a0, bh[s][0], m);
        m = wmma_f16(a1, bh[s][1], m);
        #pragma unroll
        for (int i = 0; i < 8; ++i) m[i] = m[i] > 0.f ? m[i] : 0.f;
        store_row8(sMw + s * 1024, lane, j, m);
      }
    }
    __builtin_amdgcn_wave_barrier();

    // MLP layer 2: (m @ W2 + b2) * mask, bias seeded, streamed to HBM (non-temporal)
    v16h bm[2][2];
    #pragma unroll
    for (int s = 0; s < 2; ++s) {
      bm[s][0] = load_bfrag(sMw + s * 1024, 0,  lane);
      bm[s][1] = load_bfrag(sMw + s * 1024, 32, lane);
    }
    const float mv0 = (t < nv0) ? 1.f : 0.f;
    const float mv1 = (t < nv1) ? 1.f : 0.f;
    #pragma unroll
    for (int i = 0; i < 8; ++i) {
      v16h a0 = load_afrag(sW2t, i, 0, lane), a1 = load_afrag(sW2t, i, 32, lane);
      v8f bb = load_bias8(b2s, i * 16 + sel8);
      #pragma unroll
      for (int s = 0; s < 2; ++s) {
        const float mv = s ? mv1 : mv0;
        v8f o = bb;
        o = wmma_f16(a0, bm[s][0], o);
        o = wmma_f16(a1, bm[s][1], o);
        float* op = xout + ((size_t)(rowBase + s * 16 + (lane & 15)) * NSTEP + t) * DDIM
                         + i * 16 + sel8;
        v4f lo, hi;
        #pragma unroll
        for (int k = 0; k < 4; ++k) { lo[k] = o[k] * mv; hi[k] = o[4 + k] * mv; }
        __builtin_nontemporal_store(lo, (v4f*)(op));
        __builtin_nontemporal_store(hi, (v4f*)(op + 4));
      }
    }
    // full-wave mask store: lane l covers batch row rowBase+l
    __builtin_nontemporal_store((t < nstore) ? 1.f : 0.f,
                                maskout + (size_t)(rowBase + lane) * NSTEP + t);
  }
}

extern "C" void kernel_launch(void* const* d_in, const int* in_sizes, int n_in,
                              void* d_out, int out_size, void* d_ws, size_t ws_size,
                              hipStream_t stream) {
  const float* z    = (const float*)d_in[0];
  const int*   nseq = (const int*)d_in[1];
  const float* W_ih = (const float*)d_in[2];
  const float* W_hh = (const float*)d_in[3];
  const float* b_ih = (const float*)d_in[4];
  const float* b_hh = (const float*)d_in[5];
  const float* W1   = (const float*)d_in[6];
  const float* b1   = (const float*)d_in[7];
  const float* W2   = (const float*)d_in[8];
  const float* b2   = (const float*)d_in[9];
  float* xout    = (float*)d_out;
  float* maskout = xout + (size_t)B_TOT * NSTEP * DDIM;

  dim3 grid(B_TOT / (WAVES * 32));   // 256 blocks, each = 4 waves x 32 rows
  dim3 block(WAVES * 32);
  hipLaunchKernelGGL(gru_decoder_wmma, grid, block, 0, stream,
                     z, nseq, W_ih, W_hh, b_ih, b_hh, W1, b1, W2, b2,
                     xout, maskout);
}
